// SparseActLinear_40785009442946
// MI455X (gfx1250) — compile-verified, matrix operand order checked by reading, not run on hardware
//
#include <hip/hip_runtime.h>

// bf16x3 split-precision GEMM, LDS-staged, on the gfx1250 WMMA pipe.
// M = B*S = 8192 tokens, N = OUT_F = 4096, K = IN_F = 4096.
// out[m, n] = mask[m] * (sum_k input[m,k] * weight[n,k] + bias[n])
//
// Each fp32 element is split ONCE per block into hi/lo bf16 planes in LDS
// (hi = round-half-up bf16, lo = truncated bf16 of the exact residual);
// waves read WMMA fragments with ds_load_b128 and run hi*hi + hi*lo + lo*hi.

typedef __attribute__((ext_vector_type(16))) __bf16 v16bf;
typedef __attribute__((ext_vector_type(8)))  __bf16 v8bf;
typedef __attribute__((ext_vector_type(8)))  float  v8f;

#define K_DIM 4096
#define N_DIM 4096
#define BM 256      // block tile M
#define BN 128      // block tile N
#define BK 32       // K step (one bf16 WMMA K)
#define LDSK 40     // padded LDS row stride (bf16 elems) to spread banks

// Split 8 fp32 into packed hi/lo bf16 (4 u32 each).
//  hi[e] = bf16(f[e] + half-ulp)           (round-half-up, 1 v_add per elem)
//  lo[e] = bf16_trunc(f[e] - float(hi[e])) (residual exact in fp32)
// Packing via v_perm_b32: one op per bf16 pair.
__device__ __forceinline__ void split8(const float* __restrict__ f,
                                       uint4& hi, uint4& lo) {
    unsigned h[4], l[4];
#pragma unroll
    for (int p = 0; p < 4; ++p) {
        const float f0 = f[2 * p + 0];
        const float f1 = f[2 * p + 1];
        const unsigned r0 = __float_as_uint(f0) + 0x8000u;
        const unsigned r1 = __float_as_uint(f1) + 0x8000u;
        // {r1[31:16], r0[31:16]} -> packed hi pair
        const unsigned hp = __builtin_amdgcn_perm(r1, r0, 0x07060302u);
        const float h0f = __uint_as_float(hp << 16);
        const float h1f = __uint_as_float(hp & 0xFFFF0000u);
        const float e0 = f0 - h0f;   // exact residual (v_fma_mix_f32_bf16)
        const float e1 = f1 - h1f;
        const unsigned lp = __builtin_amdgcn_perm(__float_as_uint(e1),
                                                  __float_as_uint(e0),
                                                  0x07060302u);
        h[p] = hp;
        l[p] = lp;
    }
    hi = make_uint4(h[0], h[1], h[2], h[3]);
    lo = make_uint4(l[0], l[1], l[2], l[3]);
}

__global__ __launch_bounds__(256)
void sparse_act_linear_wmma(const float* __restrict__ input,
                            const int*   __restrict__ act_mask,
                            const float* __restrict__ weight,
                            const float* __restrict__ bias,
                            float* __restrict__ out) {
    // hi plane = [0], lo plane = [1]
    __shared__ __align__(16) __bf16 sA[2][BM][LDSK];
    __shared__ __align__(16) __bf16 sB[2][BN][LDSK];

    const int t    = threadIdx.x;
    const int lane = t & 31;
    const int wave = t >> 5;
    const int wvM  = wave & 3;    // 4 waves along M
    const int wvN  = wave >> 2;   // 2 waves along N
    const int lRow = lane & 15;
    const int lHi  = lane >> 4;

    // Staging: thread t loads A row t (32 floats) and B row (t&127),
    // K half (t>>7)*16 (16 floats).
    const size_t aBase = (size_t)(blockIdx.y * BM + t) * K_DIM;
    const int    bRowL = t & 127;
    const int    bKoff = (t >> 7) * 16;
    const size_t bBase = (size_t)(blockIdx.x * BN + bRowL) * K_DIM + bKoff;

    float ra[32];
    float rb[16];

    auto loadTiles = [&](int kt) {
#pragma unroll
        for (int j = 0; j < 8; ++j)
            *reinterpret_cast<float4*>(ra + 4 * j) =
                *reinterpret_cast<const float4*>(input + aBase + kt + 4 * j);
#pragma unroll
        for (int j = 0; j < 4; ++j)
            *reinterpret_cast<float4*>(rb + 4 * j) =
                *reinterpret_cast<const float4*>(weight + bBase + kt + 4 * j);
        // Prefetch one tile AHEAD of the tile we'll load next iteration.
        __builtin_prefetch(input + aBase + kt + 2 * BK, 0, 3);  // global_prefetch_b8
        __builtin_prefetch(weight + bBase + kt + 2 * BK, 0, 3);
    };

    auto cvtStore = [&]() {
#pragma unroll
        for (int g = 0; g < 4; ++g) {           // A: 4 groups of 8
            uint4 h, l;
            split8(ra + g * 8, h, l);
            *reinterpret_cast<uint4*>(&sA[0][t][g * 8]) = h;   // ds_store_b128
            *reinterpret_cast<uint4*>(&sA[1][t][g * 8]) = l;
        }
#pragma unroll
        for (int g = 0; g < 2; ++g) {           // B: 2 groups of 8
            uint4 h, l;
            split8(rb + g * 8, h, l);
            *reinterpret_cast<uint4*>(&sB[0][bRowL][bKoff + g * 8]) = h;
            *reinterpret_cast<uint4*>(&sB[1][bRowL][bKoff + g * 8]) = l;
        }
    };

    // Fragment gather from LDS: per lane, 8 bf16 at K=lHi*8 and 8 at K=16+lHi*8
    // (matches the 16-bit A/B WMMA VGPR layout). Two ds_load_b128 each.
    auto ldFragA = [&](int plane, int row) -> v16bf {
        v8bf p0 = *reinterpret_cast<const v8bf*>(&sA[plane][row][lHi * 8]);
        v8bf p1 = *reinterpret_cast<const v8bf*>(&sA[plane][row][16 + lHi * 8]);
        return __builtin_shufflevector(p0, p1, 0, 1, 2, 3, 4, 5, 6, 7,
                                       8, 9, 10, 11, 12, 13, 14, 15);
    };
    auto ldFragB = [&](int plane, int row) -> v16bf {
        v8bf p0 = *reinterpret_cast<const v8bf*>(&sB[plane][row][lHi * 8]);
        v8bf p1 = *reinterpret_cast<const v8bf*>(&sB[plane][row][16 + lHi * 8]);
        return __builtin_shufflevector(p0, p1, 0, 1, 2, 3, 4, 5, 6, 7,
                                       8, 9, 10, 11, 12, 13, 14, 15);
    };

    v8f acc[4][4];
#pragma unroll
    for (int mi = 0; mi < 4; ++mi)
#pragma unroll
        for (int ni = 0; ni < 4; ++ni)
            acc[mi][ni] = (v8f)(0.0f);

    loadTiles(0);

    for (int kt = 0; kt < K_DIM; kt += BK) {
        cvtStore();               // convert once per element, write LDS
        __syncthreads();          // tile visible to all waves
        if (kt + BK < K_DIM)
            loadTiles(kt + BK);   // next tile's global loads fly during compute

        v16bf aH[4], aL[4];
#pragma unroll
        for (int mi = 0; mi < 4; ++mi) {
            const int row = wvM * 64 + mi * 16 + lRow;
            aH[mi] = ldFragA(0, row);
            aL[mi] = ldFragA(1, row);
        }
#pragma unroll
        for (int ni = 0; ni < 4; ++ni) {
            const int col = wvN * 64 + ni * 16 + lRow;
            const v16bf bH = ldFragB(0, col);
            const v16bf bL = ldFragB(1, col);
#pragma unroll
            for (int mi = 0; mi < 4; ++mi) {
                // bf16x3: hi*hi + hi*lo + lo*hi, f32 accumulate.
                acc[mi][ni] = __builtin_amdgcn_wmma_f32_16x16x32_bf16(
                    false, aH[mi], false, bH, (short)0, acc[mi][ni], false, false);
                acc[mi][ni] = __builtin_amdgcn_wmma_f32_16x16x32_bf16(
                    false, aH[mi], false, bL, (short)0, acc[mi][ni], false, false);
                acc[mi][ni] = __builtin_amdgcn_wmma_f32_16x16x32_bf16(
                    false, aL[mi], false, bH, (short)0, acc[mi][ni], false, false);
            }
        }
        __syncthreads();          // LDS free for next tile's stores
    }

    // Epilogue: out[m,n] = mask[m] * (acc + bias[n]).
    // D layout: lanes 0-15 -> N=lane, M=vgpr j; lanes 16-31 -> M=8+j.
    const int mW = blockIdx.y * BM + wvM * 64;
    const int nW = blockIdx.x * BN + wvN * 64;
#pragma unroll
    for (int mi = 0; mi < 4; ++mi) {
        const int mBase = mW + mi * 16 + lHi * 8;
#pragma unroll
        for (int j = 0; j < 8; ++j) {
            const int m = mBase + j;
            const float mk = (act_mask[m] != 0) ? 1.0f : 0.0f;
#pragma unroll
            for (int ni = 0; ni < 4; ++ni) {
                const int n = nW + ni * 16 + lRow;
                const float val = mk * (acc[mi][ni][j] + bias[n]);
                // Stream the 128 MB output; keep weight resident in L2.
                __builtin_nontemporal_store(val, out + (size_t)m * N_DIM + n);
            }
        }
    }
}

extern "C" void kernel_launch(void* const* d_in, const int* in_sizes, int n_in,
                              void* d_out, int out_size, void* d_ws, size_t ws_size,
                              hipStream_t stream) {
    const float* input  = (const float*)d_in[0];
    const int*   mask   = (const int*)d_in[1];
    const float* weight = (const float*)d_in[2];
    const float* bias   = (const float*)d_in[3];
    float* out = (float*)d_out;

    const int M = in_sizes[0] / K_DIM;  // 8192 tokens
    dim3 block(256);
    dim3 grid(N_DIM / BN, M / BM);      // 32 x 32 blocks
    sparse_act_linear_wmma<<<grid, block, 0, stream>>>(input, mask, weight, bias, out);
}
